// MHA_42554535968913
// MI455X (gfx1250) — compile-verified, matrix operand order checked by reading, not run on hardware
//
#include <hip/hip_runtime.h>

// ---------------------------------------------------------------------------
// MHA for MI455X (gfx1250, wave32). All GEMM work goes through
// v_wmma_f32_16x16x32_bf16. Software-pipelined with explicit ping-pong
// fragment buffers (K-loops unrolled by 2) so the double-buffer rotation
// costs zero register copies and WMMAs overlap the next step's loads.
// ---------------------------------------------------------------------------

#define DEV static __device__ __forceinline__

typedef __bf16 bf16;
typedef __attribute__((ext_vector_type(16))) __bf16 bf16x16;
typedef __attribute__((ext_vector_type(8)))  __bf16 bf16x8;
typedef __attribute__((ext_vector_type(8)))  float  f32x8;

// Problem constants
constexpr int Bn  = 2;
constexpr int Tn  = 4096;
constexpr int Dn  = 768;
constexpr int Hn  = 12;
constexpr int DKn = 64;
constexpr int Mn  = Bn * Tn;           // 8192 total rows

// ---------------------------------------------------------------------------
// Fragment helpers.
// CDNA5 16-bit A-layout (16x32, MxK): lane holds row m = lane&15; element e
// maps to k = (e>>3)*16 + (lane>>4)*8 + (e&7). A fragment is two contiguous
// 8-element (16B) chunks at column offsets c0 = kb + hi*8, c1 = kb+16+hi*8 of
// row m. B-layout (32x16, KxN) is the same pattern on the columns of B, i.e.
// two 16B chunks of row n of B^T.
// ---------------------------------------------------------------------------
DEV bf16x16 load_frag(const bf16* rowptr, int c0, int c1) {
  bf16x8 lo8 = *(const bf16x8*)(rowptr + c0);
  bf16x8 hi8 = *(const bf16x8*)(rowptr + c1);
  bf16x16 r;
#pragma unroll
  for (int i = 0; i < 8; ++i) { r[i] = lo8[i]; r[i + 8] = hi8[i]; }
  return r;
}

DEV f32x8 wmma_bf16(bf16x16 a, bf16x16 b, f32x8 c) {
  return __builtin_amdgcn_wmma_f32_16x16x32_bf16(
      /*neg_a=*/false, a, /*neg_b=*/false, b,
      /*c_mod=*/(short)0, c, /*reuse_a=*/false, /*reuse_b=*/false);
}

// 16-lane (half-wave) row reductions for the C/D layout (row r's 16 N-values
// sit on lanes 0..15 or 16..31; xor masks < 16 stay inside a half).
DEV float half_max(float v) {
#pragma unroll
  for (int m = 1; m < 16; m <<= 1) v = fmaxf(v, __shfl_xor(v, m, 32));
  return v;
}
DEV float half_sum(float v) {
#pragma unroll
  for (int m = 1; m < 16; m <<= 1) v += __shfl_xor(v, m, 32);
  return v;
}

// ---------------------------------------------------------------------------
// GEMM strip: one wave computes a 32(M) x 64(N) strip of C = A * (B^T-layout),
// K-loop by 32, ping-pong double-buffered (unroll 2).
// ---------------------------------------------------------------------------
DEV void load_set(const bf16* arow0, const bf16* arow1, const bf16* brow,
                  int ldb, int kb, int hi,
                  bf16x16& a0, bf16x16& a1, bf16x16 b[4]) {
  const int c0 = kb + hi * 8, c1 = kb + 16 + hi * 8;
  a0 = load_frag(arow0, c0, c1);
  a1 = load_frag(arow1, c0, c1);
#pragma unroll
  for (int nt = 0; nt < 4; ++nt)
    b[nt] = load_frag(brow + (size_t)nt * 16 * ldb, c0, c1);
}

DEV void mma_set(const bf16x16& a0, const bf16x16& a1, const bf16x16 b[4],
                 f32x8 c[2][4]) {
#pragma unroll
  for (int nt = 0; nt < 4; ++nt) {
    c[0][nt] = wmma_bf16(a0, b[nt], c[0][nt]);
    c[1][nt] = wmma_bf16(a1, b[nt], c[1][nt]);
  }
}

DEV void gemm_strip32(const bf16* A, int lda, int m0,
                      const bf16* Bt, int ldb, int n0,
                      int K, int lo, int hi, f32x8 c[2][4]) {
  const bf16* arow0 = A + (size_t)(m0 + lo) * lda;
  const bf16* arow1 = arow0 + (size_t)16 * lda;
  const bf16* brow  = Bt + (size_t)(n0 + lo) * ldb;

  bf16x16 a0A, a1A, bA[4];
  bf16x16 a0B, a1B, bB[4];
  load_set(arow0, arow1, brow, ldb, 0, hi, a0A, a1A, bA);

  const int nk = K / 32;                 // even (768/32 = 24)
  for (int t = 0; t < nk; t += 2) {
    const int k1 = ((t + 1 < nk) ? t + 1 : t) * 32;       // clamped
    load_set(arow0, arow1, brow, ldb, k1, hi, a0B, a1B, bB);
    mma_set(a0A, a1A, bA, c);
    const int k2 = ((t + 2 < nk) ? (t + 2) * 32 : k1);    // clamped
    load_set(arow0, arow1, brow, ldb, k2, hi, a0A, a1A, bA);
    mma_set(a0B, a1B, bB, c);
  }
}

// ---------------------------------------------------------------------------
// Kernel 1: fp32 -> bf16 conversion. x row-major; weights stored TRANSPOSED
// (Wt[n*768 + k] = W[k*768 + n]) so GEMM B-fragments are contiguous 16B loads.
// ---------------------------------------------------------------------------
__global__ void cvt_kernel(const float* __restrict__ x,
                           const float* __restrict__ WQ,
                           const float* __restrict__ WK,
                           const float* __restrict__ WV,
                           const float* __restrict__ WP,
                           bf16* __restrict__ xb,
                           bf16* __restrict__ wqT, bf16* __restrict__ wkT,
                           bf16* __restrict__ wvT, bf16* __restrict__ wpT) {
  const int stride = gridDim.x * blockDim.x;
  const int idx = blockIdx.x * blockDim.x + threadIdx.x;
  const int NX = Mn * Dn;
  for (int i = idx; i < NX; i += stride) xb[i] = (bf16)x[i];
  const int NW = Dn * Dn;
  for (int i = idx; i < NW; i += stride) {
    int kin = i / Dn, nout = i % Dn;
    int t = nout * Dn + kin;
    wqT[t] = (bf16)WQ[i];
    wkT[t] = (bf16)WK[i];
    wvT[t] = (bf16)WV[i];
    wpT[t] = (bf16)WP[i];
  }
}

// ---------------------------------------------------------------------------
// Kernel 2: QKV projection. 8 waves/block; each wave does a 32x64 strip of
// one of Q/K/V. Q is pre-scaled by DK^-1/2. Q,K stored [B,H,T,DK]; V stored
// transposed [B,H,DK,T] so the P.V B-fragments later are contiguous.
// ---------------------------------------------------------------------------
__global__ __launch_bounds__(256) void qkv_kernel(
    const bf16* __restrict__ xb,
    const bf16* __restrict__ wqT, const bf16* __restrict__ wkT,
    const bf16* __restrict__ wvT,
    bf16* __restrict__ Qo, bf16* __restrict__ Ko, bf16* __restrict__ Vt) {
  const int lane = threadIdx.x & 31;
  const int lo = lane & 15, hi = lane >> 4;
  const int w = blockIdx.x * (blockDim.x >> 5) + (threadIdx.x >> 5);

  const int per_mat = (Mn / 32) * Hn;          // 256 * 12
  const int mat = w / per_mat;                 // 0=Q 1=K 2=V
  const int rem = w % per_mat;
  const int mt = rem / Hn;                     // 32-row tile
  const int h  = rem % Hn;                     // 64-col strip == head

  const bf16* wT = (mat == 0) ? wqT : (mat == 1) ? wkT : wvT;
  f32x8 c[2][4] = {};
  gemm_strip32(xb, Dn, mt * 32, wT, Dn, h * DKn, Dn, lo, hi, c);

  const float scale = (mat == 0) ? 0.125f : 1.0f;   // DK^-0.5 folded into Q
#pragma unroll
  for (int rt = 0; rt < 2; ++rt) {
#pragma unroll
    for (int nt = 0; nt < 4; ++nt) {
#pragma unroll
      for (int vg = 0; vg < 8; ++vg) {
        int r = mt * 32 + rt * 16 + vg + 8 * hi;
        int b = r >> 12;              // / Tn
        int t = r & (Tn - 1);
        int dk = nt * 16 + lo;
        bf16 v = (bf16)(c[rt][nt][vg] * scale);
        if (mat == 2) {
          Vt[((size_t)(b * Hn + h) * DKn + dk) * Tn + t] = v;
        } else {
          bf16* dst = (mat == 0) ? Qo : Ko;
          dst[((size_t)(b * Hn + h) * Tn + t) * DKn + dk] = v;
        }
      }
    }
  }
}

// ---------------------------------------------------------------------------
// Kernel 3: flash attention. One wave per 16-query tile; 32 keys per step,
// j-loop unrolled by 2 with ping-pong K-fragment buffers. V-fragments for
// step j and K-fragments for step j+32 are issued before the softmax VALU
// section, which hides their latency.
// ---------------------------------------------------------------------------
__global__ __launch_bounds__(128) void attn_kernel(
    const bf16* __restrict__ Q, const bf16* __restrict__ K,
    const bf16* __restrict__ Vt, bf16* __restrict__ AO) {
  __shared__ bf16 ldsP[4][16][32];              // per-wave P tile (1KB each)

  const int wave = threadIdx.x >> 5;
  const int lane = threadIdx.x & 31;
  const int lo = lane & 15, hi = lane >> 4;

  const int gw = blockIdx.x * 4 + wave;         // 6144 q-tiles total
  const int qt = gw & 255;                      // % (Tn/16)
  const int bh = gw >> 8;                       // 0..23
  const int b = bh / Hn, h = bh % Hn;

  const bf16* Qh = Q  + (size_t)bh * Tn * DKn;
  const bf16* Kh = K  + (size_t)bh * Tn * DKn;
  const bf16* Vh = Vt + (size_t)bh * DKn * Tn;
  const int q0 = qt * 16;

  // Q fragments (16x64 -> two 16x32 A-fragments), resident all iterations.
  const bf16* qrow = Qh + (size_t)(q0 + lo) * DKn;
  bf16x16 qa0 = load_frag(qrow, hi * 8,      16 + hi * 8);
  bf16x16 qa1 = load_frag(qrow, 32 + hi * 8, 48 + hi * 8);

  f32x8 o[4] = {};
  float mrow[8], lrow[8];
#pragma unroll
  for (int i = 0; i < 8; ++i) { mrow[i] = -3.0e38f; lrow[i] = 0.0f; }

  // K fragments for two 16-key tiles: [tile*2 + kk-chunk]
  auto loadK = [&](int j, bf16x16 kf[4]) {
    const bf16* k0 = Kh + (size_t)(j + lo) * DKn;
    const bf16* k1 = Kh + (size_t)(j + 16 + lo) * DKn;
    kf[0] = load_frag(k0, hi * 8,      16 + hi * 8);
    kf[1] = load_frag(k0, 32 + hi * 8, 48 + hi * 8);
    kf[2] = load_frag(k1, hi * 8,      16 + hi * 8);
    kf[3] = load_frag(k1, 32 + hi * 8, 48 + hi * 8);
  };

  // One 32-key step: uses kf, prefetches the next step's K into kfn.
  auto attn_step = [&](int j, int jn, bf16x16 kf[4], bf16x16 kfn[4]) {
    // V fragments for this step (independent of softmax)
    bf16x16 vb[4];
#pragma unroll
    for (int nt = 0; nt < 4; ++nt)
      vb[nt] = load_frag(Vh + (size_t)(nt * 16 + lo) * Tn + j,
                         hi * 8, 16 + hi * 8);

    // S = Q . K^T for 32 keys (two 16x16 tiles)
    f32x8 s0 = {}, s1 = {};
    s0 = wmma_bf16(qa0, kf[0], s0);
    s0 = wmma_bf16(qa1, kf[1], s0);
    s1 = wmma_bf16(qa0, kf[2], s1);
    s1 = wmma_bf16(qa1, kf[3], s1);

    // Prefetch next step's K fragments into the other buffer.
    loadK(jn, kfn);

    // Online softmax (overlaps the loads above).
    float alpha[8];
#pragma unroll
    for (int vg = 0; vg < 8; ++vg) {
      float mx = half_max(fmaxf(s0[vg], s1[vg]));
      float mnew = fmaxf(mrow[vg], mx);
      float a = __expf(mrow[vg] - mnew);
      mrow[vg] = mnew;
      float p0 = __expf(s0[vg] - mnew);
      float p1 = __expf(s1[vg] - mnew);
      s0[vg] = p0; s1[vg] = p1;
      lrow[vg] = lrow[vg] * a + half_sum(p0 + p1);
      alpha[vg] = a;
    }
#pragma unroll
    for (int nt = 0; nt < 4; ++nt)
#pragma unroll
      for (int vg = 0; vg < 8; ++vg) o[nt][vg] *= alpha[vg];

    // P: C-layout f32 -> A-layout bf16 via LDS bounce.
#pragma unroll
    for (int vg = 0; vg < 8; ++vg) {
      ldsP[wave][vg + 8 * hi][lo]      = (bf16)s0[vg];
      ldsP[wave][vg + 8 * hi][16 + lo] = (bf16)s1[vg];
    }
    asm volatile("s_wait_dscnt 0x0" ::: "memory");  // wave-local LDS RAW
    bf16x16 pa = load_frag(&ldsP[wave][lo][0], hi * 8, 16 + hi * 8);

    // O += P . V (V already in registers).
#pragma unroll
    for (int nt = 0; nt < 4; ++nt) o[nt] = wmma_bf16(pa, vb[nt], o[nt]);
  };

  bf16x16 kfA[4], kfB[4];
  loadK(0, kfA);
  for (int j = 0; j < Tn; j += 64) {
    attn_step(j, j + 32, kfA, kfB);
    attn_step(j + 32, (j + 64 < Tn) ? j + 64 : j + 32, kfB, kfA);
  }

  // Normalize and store attn-out bf16 as row-major [B*T, 768].
  float inv[8];
#pragma unroll
  for (int vg = 0; vg < 8; ++vg) inv[vg] = 1.0f / lrow[vg];
#pragma unroll
  for (int nt = 0; nt < 4; ++nt)
#pragma unroll
    for (int vg = 0; vg < 8; ++vg) {
      int m = vg + 8 * hi;
      int t = q0 + m;
      AO[((size_t)(b * Tn + t)) * Dn + h * DKn + nt * 16 + lo] =
          (bf16)(o[nt][vg] * inv[vg]);
    }
}

// ---------------------------------------------------------------------------
// Kernel 4: output projection (32x64 strips), fp32 result straight to d_out.
// ---------------------------------------------------------------------------
__global__ __launch_bounds__(256) void proj_kernel(
    const bf16* __restrict__ AO, const bf16* __restrict__ wpT,
    float* __restrict__ out) {
  const int lane = threadIdx.x & 31;
  const int lo = lane & 15, hi = lane >> 4;
  const int w = blockIdx.x * (blockDim.x >> 5) + (threadIdx.x >> 5);
  const int mt = w / (Dn / DKn);                // 32-row tile
  const int ns = w % (Dn / DKn);                // 64-col strip

  f32x8 c[2][4] = {};
  gemm_strip32(AO, Dn, mt * 32, wpT, Dn, ns * DKn, Dn, lo, hi, c);

#pragma unroll
  for (int rt = 0; rt < 2; ++rt)
#pragma unroll
    for (int nt = 0; nt < 4; ++nt)
#pragma unroll
      for (int vg = 0; vg < 8; ++vg) {
        int r = mt * 32 + rt * 16 + vg + 8 * hi;
        out[(size_t)r * Dn + ns * DKn + nt * 16 + lo] = c[rt][nt][vg];
      }
}

// ---------------------------------------------------------------------------
extern "C" void kernel_launch(void* const* d_in, const int* in_sizes, int n_in,
                              void* d_out, int out_size, void* d_ws,
                              size_t ws_size, hipStream_t stream) {
  (void)in_sizes; (void)n_in; (void)out_size; (void)ws_size;
  const float* x  = (const float*)d_in[0];
  const float* WQ = (const float*)d_in[1];
  const float* WK = (const float*)d_in[2];
  const float* WV = (const float*)d_in[3];
  const float* WP = (const float*)d_in[4];

  const size_t NX = (size_t)Mn * Dn;     // 6,291,456
  const size_t NW = (size_t)Dn * Dn;     //   589,824
  bf16* xb  = (bf16*)d_ws;
  bf16* wqT = xb  + NX;
  bf16* wkT = wqT + NW;
  bf16* wvT = wkT + NW;
  bf16* wpT = wvT + NW;
  bf16* Qb  = wpT + NW;
  bf16* Kb  = Qb  + NX;
  bf16* Vt  = Kb  + NX;
  bf16* AO  = Vt  + NX;                  // total ~67.6 MB of workspace

  cvt_kernel<<<2048, 256, 0, stream>>>(x, WQ, WK, WV, WP,
                                       xb, wqT, wkT, wvT, wpT);

  // 3 mats * 256 row-tiles * 12 strips = 9216 waves / 8 per block
  qkv_kernel<<<1152, 256, 0, stream>>>(xb, wqT, wkT, wvT, Qb, Kb, Vt);

  // 24 heads * 256 q-tiles = 6144 waves / 4 per block
  attn_kernel<<<1536, 128, 0, stream>>>(Qb, Kb, Vt, AO);

  // 256 row-tiles * 12 strips = 3072 waves / 8 per block
  proj_kernel<<<384, 256, 0, stream>>>(AO, wpT, (float*)d_out);
}
